// LSTMClassifier_55405078118420
// MI455X (gfx1250) — matrix operand order repaired and checked by reading, not compiled, not run on hardware
//
#include <hip/hip_runtime.h>
#include <hip/hip_bf16.h>

// ---------------------------------------------------------------------------
// 2-layer LayerNorm-LSTM classifier for MI455X (gfx1250, wave32, WMMA).
// bf16 weights transposed to [N,K] (K-contiguous) stream from L2;
// V_WMMA_F32_16X16X32_BF16 with fp32 accumulation.
// GEMM tiling: one wave owns the full M=64 strip of one N tile (4 acc tiles,
// 4 WMMA per B-fragment) => recurrent-weight L2 traffic at the 25MB/step
// floor. A (h-state) is DMA'd into a double-buffered LDS tile with
// GLOBAL_LOAD_ASYNC_TO_LDS_B128 (ASYNCcnt) so the next chunk's transfer
// overlaps the current chunk's WMMAs.
// ---------------------------------------------------------------------------

typedef __bf16 bf16_t;
typedef __attribute__((ext_vector_type(16))) __bf16 v16bf;
typedef __attribute__((ext_vector_type(8)))  __bf16 v8bf;
typedef __attribute__((ext_vector_type(8)))  float  v8f;

#define B_     64
#define T_     512
#define F_IN_  128
#define UNITS_ 1024
#define GATES_ 4096
#define NC_    16
#define APAD_  136   // 128-elem K chunk + 8-elem pad: rows 272B apart -> 4-bank shift
#define ABUF_  (64 * APAD_)

__device__ __forceinline__ v16bf cat16(v8bf lo, v8bf hi) {
  v16bf r;
#pragma unroll
  for (int i = 0; i < 8; ++i) { r[i] = lo[i]; r[i + 8] = hi[i]; }
  return r;
}

__device__ __forceinline__ float sigmoidf_(float x) {
  return 1.0f / (1.0f + __expf(-x));
}

// Async-DMA one 64x128 bf16 A chunk into LDS (padded row stride APAD_).
// Each thread issues 4 b128 lane-transfers; tracked on ASYNCcnt.
__device__ __forceinline__ void issue_async_chunk(const bf16_t* __restrict__ A,
                                                  long a_stride, int k0,
                                                  bf16_t* dst, int tid) {
#pragma unroll
  for (int i = 0; i < 4; ++i) {
    int idx = tid + i * 256;                 // 0..1023
    int row = idx >> 4;                      // 0..63
    int c8  = idx & 15;                      // 16 chunks of 8 bf16
    const bf16_t* gp = A + (long)row * a_stride + k0 + c8 * 8;
    unsigned lds = (unsigned)(unsigned long long)(dst + row * APAD_ + c8 * 8);
    asm volatile("global_load_async_to_lds_b128 %0, %1, off"
                 :: "v"(lds), "v"(gp) : "memory");
  }
}

// Accumulate A[0:64, 0:K) x B^T-row (this lane's N column) into 4 M-tiles,
// double-buffering A chunks through LDS via the async-DMA path.
// Operand layouts per ISA:
//  A 16-bit 16x32: lanes0-15 hold K={0..7,16..23}, lanes16-31 K={8..15,24..31}
//  B 16-bit 32x16: lanes0-15 hold K={0..7,16..23}, lanes16-31 K={8..15,24..31}
__device__ __forceinline__ void gemm_async(const bf16_t* __restrict__ A, long a_stride,
                                           const bf16_t* __restrict__ Brow, int K,
                                           bf16_t* Atile, int tid, int lr, int g,
                                           v8f acc[4]) {
  const int nch = K >> 7;                    // K / 128 chunks
  issue_async_chunk(A, a_stride, 0, Atile, tid);
  int buf = 0;
  for (int c = 0; c < nch; ++c) {
    if (c + 1 < nch) {
      issue_async_chunk(A, a_stride, (c + 1) << 7, Atile + (buf ^ 1) * ABUF_, tid);
      asm volatile("s_wait_asynccnt 0x4" ::: "memory");   // chunk c done (in-order)
    } else {
      asm volatile("s_wait_asynccnt 0x0" ::: "memory");
    }
    __syncthreads();                         // all waves' chunk-c DMA visible
    const bf16_t* Abuf = Atile + buf * ABUF_;
#pragma unroll
    for (int kk = 0; kk < 128; kk += 32) {
      int k = (c << 7) + kk;
      __builtin_prefetch(Brow + k + 512, 0, 3);           // stream B through caches
      v16bf bfrag = cat16(*(const v8bf*)(Brow + k + g * 8),
                          *(const v8bf*)(Brow + k + 16 + g * 8));
      v16bf af[4];
#pragma unroll
      for (int m = 0; m < 4; ++m) {          // preload all A frags: one ds wait
        const bf16_t* ar = Abuf + (m * 16 + lr) * APAD_ + kk;
        af[m] = cat16(*(const v8bf*)(ar + g * 8),
                      *(const v8bf*)(ar + 16 + g * 8));
      }
#pragma unroll
      for (int m = 0; m < 4; ++m) {          // 4 back-to-back WMMAs per B frag
        acc[m] = __builtin_amdgcn_wmma_f32_16x16x32_bf16(false, af[m], false, bfrag,
                                                         (short)0, acc[m], false, false);
      }
    }
    __syncthreads();                         // buf free for re-DMA
    buf ^= 1;
  }
}

// z[64,4096] = A1[64,K1] @ Bt1^T + A2[64,K2] @ Bt2^T + bias
// Bt*: weights pre-transposed to [4096, K] bf16 (rows K-contiguous).
// Grid: 32 blocks x 256 thr (8 waves); wave -> one N tile, all 4 M tiles.
__global__ void lstm_gemm(const bf16_t* __restrict__ A1, long a1_stride, int K1,
                          const bf16_t* __restrict__ A2, int K2,
                          const bf16_t* __restrict__ Bt1,
                          const bf16_t* __restrict__ Bt2,
                          const float*  __restrict__ bias,
                          float* __restrict__ z) {
  __shared__ __align__(16) bf16_t Atile[2 * ABUF_];
  int tid  = threadIdx.x;
  int lane = tid & 31;
  int wave = tid >> 5;
  int nt   = blockIdx.x * 8 + wave;          // 0..255
  int g    = lane >> 4;
  int lr   = lane & 15;
  int n    = nt * 16 + lr;                   // N column for B / C / D

  v8f acc[4] = {};
  gemm_async(A1, a1_stride, Bt1 + (long)n * K1, K1, Atile, tid, lr, g, acc);
  gemm_async(A2, (long)K2,  Bt2 + (long)n * K2, K2, Atile, tid, lr, g, acc);

  float bv = bias[n];
  // C/D layout: lanes0-15 -> M = m*16 + r, lanes16-31 -> M = m*16 + 8 + r
#pragma unroll
  for (int m = 0; m < 4; ++m) {
#pragma unroll
    for (int r = 0; r < 8; ++r) {
      int row = m * 16 + g * 8 + r;
      z[(long)row * GATES_ + n] = acc[m][r] + bv;
    }
  }
}

// Gate nonlinearities + cell/hidden update + LayerNorm; optionally FC+softmax.
// Grid: 64 blocks (one per batch row) x 256 threads (4 units/thread).
__global__ void lstm_gates_ln(const float* __restrict__ z,     // [64,4096]
                              float* __restrict__ h,           // [64,1024]
                              float* __restrict__ c,           // [64,1024]
                              bf16_t* __restrict__ h_b,        // bf16 h (next step A)
                              bf16_t* __restrict__ o_b,        // bf16 LN out (next layer A)
                              const float* __restrict__ gamma,
                              const float* __restrict__ beta,
                              const float* __restrict__ Wfc,   // [1024,16]
                              const float* __restrict__ bfc,   // [16]
                              float* __restrict__ out,         // [64,512,16]
                              int t, int do_logits) {
  __shared__ float ln_y[UNITS_];
  __shared__ float rs[8], rss[8];
  __shared__ float smean, srstd;
  __shared__ float pl[256];
  __shared__ float logits[NC_];

  int b   = blockIdx.x;
  int tid = threadIdx.x;
  const float* zb = z + (long)b * GATES_;
  float* cb = c + (long)b * UNITS_;
  float* hb = h + (long)b * UNITS_;

  float hv[4];
  float s = 0.f, ss = 0.f;
#pragma unroll
  for (int j = 0; j < 4; ++j) {
    int u = tid + j * 256;
    float iv = sigmoidf_(zb[u]);
    float fv = sigmoidf_(zb[UNITS_ + u]);
    float gv = tanhf(zb[2 * UNITS_ + u]);
    float ov = sigmoidf_(zb[3 * UNITS_ + u]);
    float cn = fv * cb[u] + iv * gv;
    float hn = ov * tanhf(cn);
    cb[u] = cn;
    hb[u] = hn;
    h_b[(long)b * UNITS_ + u] = (bf16_t)hn;
    hv[j] = hn;
    s  += hn;
    ss += hn * hn;
  }

  // wave32 shuffle reduction, then cross-wave via LDS
#pragma unroll
  for (int off = 16; off > 0; off >>= 1) {
    s  += __shfl_down(s, off);
    ss += __shfl_down(ss, off);
  }
  if ((tid & 31) == 0) { rs[tid >> 5] = s; rss[tid >> 5] = ss; }
  __syncthreads();
  if (tid == 0) {
    float ts = 0.f, tss = 0.f;
#pragma unroll
    for (int w = 0; w < 8; ++w) { ts += rs[w]; tss += rss[w]; }
    float mean = ts * (1.0f / UNITS_);
    float var  = tss * (1.0f / UNITS_) - mean * mean;
    smean = mean;
    srstd = rsqrtf(var + 1e-3f);
  }
  __syncthreads();
  float mean = smean, rstd = srstd;

#pragma unroll
  for (int j = 0; j < 4; ++j) {
    int u = tid + j * 256;
    float y = (hv[j] - mean) * rstd * gamma[u] + beta[u];
    o_b[(long)b * UNITS_ + u] = (bf16_t)y;
    ln_y[u] = y;
  }
  __syncthreads();

  if (do_logits) {
    // FC: 16 classes x 16 chunks of 64 units each
    int cls   = tid & 15;
    int chunk = tid >> 4;
    float p = 0.f;
    const float* wc = Wfc + cls;
#pragma unroll 4
    for (int e = 0; e < 64; ++e) {
      int u = chunk * 64 + e;
      p += ln_y[u] * wc[u * NC_];
    }
    pl[tid] = p;
  }
  __syncthreads();
  if (do_logits && tid < NC_) {
    float l = bfc[tid];
#pragma unroll
    for (int ch = 0; ch < 16; ++ch) l += pl[ch * 16 + tid];
    logits[tid] = l;
  }
  __syncthreads();
  if (do_logits && tid < NC_) {
    float mx = logits[0];
#pragma unroll
    for (int i = 1; i < NC_; ++i) mx = fmaxf(mx, logits[i]);
    float sum = 0.f;
#pragma unroll
    for (int i = 0; i < NC_; ++i) sum += __expf(logits[i] - mx);
    out[((long)b * T_ + t) * NC_ + tid] = __expf(logits[tid] - mx) / sum;
  }
}

// W[K,4096] fp32 -> Wt[4096,K] bf16 (output-coalesced)
__global__ void conv_transpose_bf16(const float* __restrict__ W,
                                    bf16_t* __restrict__ Wt, int K) {
  long idx = (long)blockIdx.x * blockDim.x + threadIdx.x;  // over 4096*K
  int  k = (int)(idx % K);
  long n = idx / K;
  Wt[idx] = (bf16_t)W[(long)k * GATES_ + n];
}

__global__ void conv_bf16(const float* __restrict__ x, bf16_t* __restrict__ y) {
  long idx = (long)blockIdx.x * blockDim.x + threadIdx.x;
  y[idx] = (bf16_t)x[idx];
}

extern "C" void kernel_launch(void* const* d_in, const int* in_sizes, int n_in,
                              void* d_out, int out_size, void* d_ws, size_t ws_size,
                              hipStream_t stream) {
  const float* inputs = (const float*)d_in[0];
  const float* W0  = (const float*)d_in[1];
  const float* U0  = (const float*)d_in[2];
  const float* b0  = (const float*)d_in[3];
  const float* g0  = (const float*)d_in[4];
  const float* be0 = (const float*)d_in[5];
  const float* W1  = (const float*)d_in[6];
  const float* U1  = (const float*)d_in[7];
  const float* b1  = (const float*)d_in[8];
  const float* g1  = (const float*)d_in[9];
  const float* be1 = (const float*)d_in[10];
  const float* Wfc = (const float*)d_in[11];
  const float* bfc = (const float*)d_in[12];
  float* out = (float*)d_out;

  // ---- workspace layout (~40 MB) ----
  char* ws = (char*)d_ws;
  size_t o = 0;
  bf16_t* Xb  = (bf16_t*)(ws + o); o += (size_t)B_ * T_ * F_IN_ * 2;     // 8 MB
  bf16_t* Wt0 = (bf16_t*)(ws + o); o += (size_t)GATES_ * F_IN_ * 2;      // 1 MB
  bf16_t* Ut0 = (bf16_t*)(ws + o); o += (size_t)GATES_ * UNITS_ * 2;     // 8 MB
  bf16_t* Wt1 = (bf16_t*)(ws + o); o += (size_t)GATES_ * UNITS_ * 2;     // 8 MB
  bf16_t* Ut1 = (bf16_t*)(ws + o); o += (size_t)GATES_ * UNITS_ * 2;     // 8 MB
  // zeroed-state region (contiguous: single memset)
  char* zero_base = ws + o;
  float*  h0f = (float*)(ws + o);  o += (size_t)B_ * UNITS_ * 4;
  float*  c0f = (float*)(ws + o);  o += (size_t)B_ * UNITS_ * 4;
  float*  h1f = (float*)(ws + o);  o += (size_t)B_ * UNITS_ * 4;
  float*  c1f = (float*)(ws + o);  o += (size_t)B_ * UNITS_ * 4;
  bf16_t* h0b = (bf16_t*)(ws + o); o += (size_t)B_ * UNITS_ * 2;
  bf16_t* h1b = (bf16_t*)(ws + o); o += (size_t)B_ * UNITS_ * 2;
  size_t zero_bytes = (size_t)(ws + o - zero_base);
  bf16_t* o0b = (bf16_t*)(ws + o); o += (size_t)B_ * UNITS_ * 2;
  bf16_t* o1b = (bf16_t*)(ws + o); o += (size_t)B_ * UNITS_ * 2;
  float*  z0  = (float*)(ws + o);  o += (size_t)B_ * GATES_ * 4;         // 1 MB
  float*  z1  = (float*)(ws + o);  o += (size_t)B_ * GATES_ * 4;         // 1 MB
  (void)ws_size; (void)in_sizes; (void)n_in; (void)out_size;

  // ---- one-time prep per call (deterministic) ----
  conv_bf16<<<(B_ * T_ * F_IN_) / 256, 256, 0, stream>>>(inputs, Xb);
  conv_transpose_bf16<<<(GATES_ * F_IN_) / 256, 256, 0, stream>>>(W0, Wt0, F_IN_);
  conv_transpose_bf16<<<(GATES_ * UNITS_) / 256, 256, 0, stream>>>(U0, Ut0, UNITS_);
  conv_transpose_bf16<<<(GATES_ * UNITS_) / 256, 256, 0, stream>>>(W1, Wt1, UNITS_);
  conv_transpose_bf16<<<(GATES_ * UNITS_) / 256, 256, 0, stream>>>(U1, Ut1, UNITS_);
  hipMemsetAsync(zero_base, 0, zero_bytes, stream);

  // ---- sequential recurrence over T ----
  for (int t = 0; t < T_; ++t) {
    // layer 0: z0 = x_t @ W0 + h0 @ U0 + b0
    lstm_gemm<<<32, 256, 0, stream>>>(Xb + (long)t * F_IN_, (long)T_ * F_IN_, F_IN_,
                                      h0b, UNITS_, Wt0, Ut0, b0, z0);
    lstm_gates_ln<<<64, 256, 0, stream>>>(z0, h0f, c0f, h0b, o0b, g0, be0,
                                          Wfc, bfc, out, t, /*do_logits=*/0);
    // layer 1: z1 = o0 @ W1 + h1 @ U1 + b1
    lstm_gemm<<<32, 256, 0, stream>>>(o0b, UNITS_, UNITS_,
                                      h1b, UNITS_, Wt1, Ut1, b1, z1);
    lstm_gates_ln<<<64, 256, 0, stream>>>(z1, h1f, c1f, h1b, o1b, g1, be1,
                                          Wfc, bfc, out, t, /*do_logits=*/1);
  }
}
